// SelfAttention_13073880449267
// MI455X (gfx1250) — matrix operand order repaired and checked by reading, not compiled, not run on hardware
//
#include <hip/hip_runtime.h>

typedef __attribute__((ext_vector_type(16))) __bf16 v16bf;
typedef __attribute__((ext_vector_type(8)))  float  v8f;
typedef unsigned short u16;
typedef unsigned int   u32;

#define B_  16
#define C_  128
#define K8  16
#define C2  64
#define N_  4096
#define M_  1024

__device__ __forceinline__ u16 f2bf(float f) {
  u32 u = __float_as_uint(f);
  u += 0x7fffu + ((u >> 16) & 1u);        // round-to-nearest-even
  return (u16)(u >> 16);
}
__device__ __forceinline__ float bf2f(u16 h) {
  return __uint_as_float(((u32)h) << 16);
}

union BF16V {
  v16bf v;
  uint4 q[2];
};

__device__ __forceinline__ v8f wmma_bf16(v16bf a, v16bf b, v8f c) {
  // D = A(16x32 bf16) * B(32x16 bf16) + C(16x16 f32)
  return __builtin_amdgcn_wmma_f32_16x16x32_bf16(
      false, a, false, b, (short)0, c, false, false);
}

// ---------------------------------------------------------------------------
// Kernel 1: all three 1x1 convs in one pass over x (read x once).
// Thread = one (b, n) pixel, computes 16 theta + 16 phi + 64 g outputs.
// Weights transposed in LDS -> float4 broadcast reads.
// ---------------------------------------------------------------------------
__global__ __launch_bounds__(256) void k_conv(
    const float* __restrict__ x,
    const float* __restrict__ wt, const float* __restrict__ bt,
    const float* __restrict__ wp, const float* __restrict__ bp,
    const float* __restrict__ wg, const float* __restrict__ bg,
    const float* __restrict__ wa,
    u16* __restrict__ theta_bf, u16* __restrict__ tmp_phi,
    u16* __restrict__ tmp_g, u16* __restrict__ wattn_bf)
{
  __shared__ float sw[C_ * 96];   // transposed: sw[c*96 + j], j: 0..15 th, 16..31 ph, 32..95 g
  __shared__ float sb[96];
  const int tid = threadIdx.x;
  for (int i = tid; i < 96 * C_; i += 256) {
    int row = i >> 7, c = i & 127;
    float v = (row < 16) ? wt[row * C_ + c]
            : (row < 32) ? wp[(row - 16) * C_ + c]
                         : wg[(row - 32) * C_ + c];
    sw[c * 96 + row] = v;
  }
  if (tid < 96)
    sb[tid] = (tid < 16) ? bt[tid] : (tid < 32) ? bp[tid - 16] : bg[tid - 32];
  __syncthreads();

  const int gid = blockIdx.x * 256 + tid;
  if (gid < C_ * C2) wattn_bf[gid] = f2bf(wa[gid]);   // bf16 copy of w_attn
  const int b = gid >> 12;
  const int n = gid & (N_ - 1);

  float acc[96];
#pragma unroll
  for (int j = 0; j < 96; ++j) acc[j] = 0.f;

  const float* xp = x + (size_t)b * C_ * N_ + n;
  for (int c = 0; c < C_; ++c) {
    float xv = xp[(size_t)c * N_];
    const float4* wr = (const float4*)&sw[c * 96];
#pragma unroll
    for (int j4 = 0; j4 < 24; ++j4) {
      float4 w = wr[j4];
      acc[4 * j4 + 0] += w.x * xv;
      acc[4 * j4 + 1] += w.y * xv;
      acc[4 * j4 + 2] += w.z * xv;
      acc[4 * j4 + 3] += w.w * xv;
    }
  }
#pragma unroll
  for (int j = 0; j < 96; ++j) acc[j] += sb[j];

  u16* tb = theta_bf + (size_t)b * K8 * N_ + n;  // [b][k][n]
  u16* pb = tmp_phi  + (size_t)b * K8 * N_ + n;  // [b][k][n] per-pixel
#pragma unroll
  for (int k = 0; k < 16; ++k) {
    tb[(size_t)k * N_] = f2bf(acc[k]);
    pb[(size_t)k * N_] = f2bf(acc[16 + k]);
  }
  u16* gb = tmp_g + (size_t)b * C2 * N_ + n;     // [b][c][n] per-pixel
#pragma unroll
  for (int c = 0; c < 64; ++c) gb[(size_t)c * N_] = f2bf(acc[32 + c]);
}

// ---------------------------------------------------------------------------
// Kernel 2: 2x2 max-pool of per-pixel phi/g into WMMA-friendly layouts.
// phi -> [b][m][16] (A-operand rows), g -> [b][c][m] (A-operand rows)
// ---------------------------------------------------------------------------
__global__ __launch_bounds__(256) void k_pool(
    const u16* __restrict__ tmp_phi, const u16* __restrict__ tmp_g,
    u16* __restrict__ phi_bf, u16* __restrict__ g_bf)
{
  const int gid = blockIdx.x * 256 + threadIdx.x;
  if (gid >= B_ * 80 * M_) return;
  const int m  = gid & (M_ - 1);
  const int ch = (gid >> 10) % 80;
  const int b  = gid / (80 * M_);
  const int my = m >> 5, mx = m & 31;
  const int n0 = (my * 2) * 64 + mx * 2;
  const u16* src = (ch < 16)
      ? tmp_phi + ((size_t)b * K8 + ch) * N_
      : tmp_g  + ((size_t)b * C2 + (ch - 16)) * N_;
  float v = fmaxf(fmaxf(bf2f(src[n0]),      bf2f(src[n0 + 1])),
                  fmaxf(bf2f(src[n0 + 64]), bf2f(src[n0 + 65])));
  if (ch < 16) phi_bf[((size_t)b * M_ + m) * K8 + ch] = f2bf(v);
  else         g_bf[((size_t)b * C2 + (ch - 16)) * M_ + m] = f2bf(v);
}

// ---------------------------------------------------------------------------
// Kernel 3: fused flash-style attention + output projection.
// 8 waves/block; each wave owns one 16-row n-tile of one batch.
// Sgemm form: S^T tile = phi(16m x K16pad32) x theta(K x 16n) so softmax-M
// reduction runs along accumulator VGPRs (+1 shfl_xor across half-waves).
// ---------------------------------------------------------------------------
__global__ __launch_bounds__(256) void k_attn(
    const u16* __restrict__ theta_bf, const u16* __restrict__ phi_bf,
    const u16* __restrict__ g_bf, const u16* __restrict__ wattn_bf,
    const float* __restrict__ b_attn, const float* __restrict__ sigma,
    const float* __restrict__ x, float* __restrict__ out)
{
  __shared__ __align__(32) u16 sP[8 * 32 * 16];   // per-wave P-transpose scratch
  __shared__ __align__(32) u16 sO[8 * 64 * 16];   // per-wave O-transpose scratch

  const int lane = threadIdx.x & 31;
  const int wv   = threadIdx.x >> 5;
  const int half = lane >> 4;      // 0: lanes 0-15, 1: lanes 16-31
  const int l16  = lane & 15;
  const int b    = blockIdx.y;
  const int n0   = (blockIdx.x * 8 + wv) * 16;

  const v8f vzero = {};

  // Loop-invariant B operand: theta cols n0..n0+15, K=16 zero-padded to 32.
  // B layout: lane col n = l16; elems 0..7 = K half*8..+7; elems 8..15 (K=16..31) padding -> zero.
  BF16V bth;
  bth.q[1] = make_uint4(0, 0, 0, 0);
  {
    const u16* tp = theta_bf + ((size_t)b * K8 + half * 8) * N_ + n0 + l16;
    u32 w[4];
    for (int j = 0; j < 4; ++j)
      w[j] = (u32)tp[(size_t)(2 * j) * N_] | ((u32)tp[(size_t)(2 * j + 1) * N_] << 16);
    bth.q[0] = make_uint4(w[0], w[1], w[2], w[3]);
  }

  // A operand source for phi: lane row = m0 + l16, K span half*8..half*8+7.
  const u16* phi_b = phi_bf + (size_t)b * M_ * K8 + (size_t)l16 * K8 + half * 8;

  // ---- pass 1: row max over all M=1024 ----
  float vmax = -3.402823466e38f;
  for (int mt = 0; mt < 64; ++mt) {
    BF16V aph;
    aph.q[1] = make_uint4(0, 0, 0, 0);            // K=16..31 padding
    aph.q[0] = *(const uint4*)(phi_b + (size_t)mt * 16 * K8);
    v8f s = wmma_bf16(aph.v, bth.v, vzero);
#pragma unroll
    for (int i = 0; i < 8; ++i) vmax = fmaxf(vmax, s[i]);
  }
  vmax = fmaxf(vmax, __shfl_xor(vmax, 16, 32));   // partner lane shares column n

  // ---- pass 2: recompute S, exp, sum, accumulate O = g @ P ----
  float vsum = 0.f;
  v8f Oacc[4];
#pragma unroll
  for (int t = 0; t < 4; ++t) Oacc[t] = vzero;
  u16* myP = sP + wv * (32 * 16);
  const u16* g_base = g_bf + (size_t)b * C2 * M_;

  for (int mc = 0; mc < 32; ++mc) {               // 32-m chunks
#pragma unroll
    for (int h2 = 0; h2 < 2; ++h2) {
      const int mt = mc * 2 + h2;
      BF16V aph;
      aph.q[1] = make_uint4(0, 0, 0, 0);
      aph.q[0] = *(const uint4*)(phi_b + (size_t)mt * 16 * K8);
      v8f s = wmma_bf16(aph.v, bth.v, vzero);
#pragma unroll
      for (int i = 0; i < 8; ++i) {
        float p = __expf(s[i] - vmax);
        vsum += p;
        // D layout: lane col n=l16, row m-offset = half*8 + i
        myP[l16 * 32 + (h2 * 16 + half * 8 + i)] = f2bf(p);
      }
    }
    asm volatile("s_wait_dscnt 0" ::: "memory");  // P visible (in-order LDS, same wave)
    BF16V bp;                                     // B layout: lane = n, elems = K(=m offset)
    bp.q[0] = *(const uint4*)&myP[l16 * 32 + half * 8];
    bp.q[1] = *(const uint4*)&myP[l16 * 32 + 16 + half * 8];
#pragma unroll
    for (int t = 0; t < 4; ++t) {                 // 64 g-channels = 4 c-tiles
      const u16* gp = g_base + (size_t)(t * 16 + l16) * M_ + mc * 32 + half * 8;
      BF16V ag;                                   // A layout K=32: elems 0..7 / 8..15
      ag.q[0] = *(const uint4*)gp;
      ag.q[1] = *(const uint4*)(gp + 16);
      Oacc[t] = wmma_bf16(ag.v, bp.v, Oacc[t]);
    }
  }
  vsum += __shfl_xor(vsum, 16, 32);
  const float rinv = 1.0f / vsum;

  // ---- normalize O, stage through LDS as B operand for w_attn matmul ----
  u16* myO = sO + wv * (64 * 16);
#pragma unroll
  for (int t = 0; t < 4; ++t) {
#pragma unroll
    for (int i = 0; i < 8; ++i) {
      int c2 = t * 16 + half * 8 + i;
      myO[l16 * 64 + c2] = f2bf(Oacc[t][i] * rinv);
    }
  }
  asm volatile("s_wait_dscnt 0" ::: "memory");
  BF16V bo0, bo1;                                 // K-chunks c2=0..31, 32..63
  bo0.q[0] = *(const uint4*)&myO[l16 * 64 + half * 8];
  bo0.q[1] = *(const uint4*)&myO[l16 * 64 + 16 + half * 8];
  bo1.q[0] = *(const uint4*)&myO[l16 * 64 + 32 + half * 8];
  bo1.q[1] = *(const uint4*)&myO[l16 * 64 + 48 + half * 8];

  // ---- final projection: out = x + sigma * (w_attn @ O + b_attn) ----
  const float sg = sigma[0];
  const float* xb = x   + (size_t)b * C_ * N_ + n0 + l16;
  float*       ob = out + (size_t)b * C_ * N_ + n0 + l16;

#pragma unroll
  for (int tt = 0; tt < 8; ++tt) {                // 128 out-channels = 8 tiles
    v8f y = vzero;
    const u16* wpt = wattn_bf + (size_t)(tt * 16 + l16) * C2 + half * 8;
    BF16V aw;
    aw.q[0] = *(const uint4*)wpt;                 // c2 chunk 0..31
    aw.q[1] = *(const uint4*)(wpt + 16);
    y = wmma_bf16(aw.v, bo0.v, y);
    aw.q[0] = *(const uint4*)(wpt + 32);          // c2 chunk 32..63
    aw.q[1] = *(const uint4*)(wpt + 48);
    y = wmma_bf16(aw.v, bo1.v, y);
#pragma unroll
    for (int i = 0; i < 8; ++i) {
      int cout = tt * 16 + half * 8 + i;
      size_t idx = (size_t)cout * N_;
      ob[idx] = xb[idx] + sg * (y[i] + b_attn[cout]);
    }
  }
}

// ---------------------------------------------------------------------------
extern "C" void kernel_launch(void* const* d_in, const int* in_sizes, int n_in,
                              void* d_out, int out_size, void* d_ws, size_t ws_size,
                              hipStream_t stream) {
  const float* x  = (const float*)d_in[0];
  const float* wt = (const float*)d_in[1];
  const float* bt = (const float*)d_in[2];
  const float* wp = (const float*)d_in[3];
  const float* bp = (const float*)d_in[4];
  const float* wg = (const float*)d_in[5];
  const float* bg = (const float*)d_in[6];
  const float* wa = (const float*)d_in[7];
  const float* ba = (const float*)d_in[8];
  const float* sg = (const float*)d_in[9];
  float* out = (float*)d_out;

  char* ws = (char*)d_ws;
  u16* theta_bf = (u16*)(ws);                               // 2 MB   [b][k][n]
  u16* phi_bf   = (u16*)(ws + (2u << 20));                  // 512 KB [b][m][k]
  u16* g_bf     = (u16*)(ws + (2u << 20) + (512u << 10));   // 2 MB   [b][c][m]
  u16* wattn_bf = (u16*)(ws + (4u << 20) + (512u << 10));   // 16 KB

  // Big per-pixel temps live inside d_out (10 MB of 32 MB); k_attn fully
  // overwrites d_out afterwards, so this is deterministic and safe.
  u16* tmp_phi = (u16*)d_out;                               // 2 MB
  u16* tmp_g   = (u16*)d_out + (size_t)B_ * K8 * N_;        // 8 MB

  k_conv<<<(B_ * N_) / 256, 256, 0, stream>>>(x, wt, bt, wp, bp, wg, bg, wa,
                                              theta_bf, tmp_phi, tmp_g, wattn_bf);
  k_pool<<<(B_ * 80 * M_) / 256, 256, 0, stream>>>(tmp_phi, tmp_g, phi_bf, g_bf);
  k_attn<<<dim3(N_ / (16 * 8), B_), 256, 0, stream>>>(theta_bf, phi_bf, g_bf,
                                                      wattn_bf, ba, sg, x, out);
}
